// GraphAttnModel_89421219103180
// MI455X (gfx1250) — matrix operand order repaired and checked by reading, not compiled, not run on hardware
//
#include <hip/hip_runtime.h>
#include <hip/hip_bf16.h>
#include <hip/hip_fp16.h>

typedef __attribute__((ext_vector_type(16))) _Float16 v16h;
typedef __attribute__((ext_vector_type(8)))  _Float16 v8h;
typedef __attribute__((ext_vector_type(8)))  float    v8f;

#define F_IN 256     // input features
#define HD   256     // H*D
#define NH   4       // heads
#define KOUT 1024    // q|k|v|skip concatenated output width

// ---------------------------------------------------------------- helpers ---
__device__ __forceinline__ void atomicMaxFloat(float* addr, float val) {
    // standard sign-punning trick; addr initialized to -1e30f
    if (val >= 0.0f) {
        atomicMax((int*)addr, __float_as_int(val));
    } else {
        atomicMin((unsigned int*)addr, (unsigned int)__float_as_int(val));
    }
}

__device__ __forceinline__ float blockReduceSum256(float v, float* smem) {
    // wave32 in-register reduce, then 8 wave partials summed by everyone
    #pragma unroll
    for (int m = 16; m >= 1; m >>= 1) v += __shfl_xor(v, m, 32);
    const int lane = threadIdx.x & 31;
    const int w    = threadIdx.x >> 5;
    if (lane == 0) smem[w] = v;
    __syncthreads();
    float t = 0.0f;
    #pragma unroll
    for (int i = 0; i < 8; i++) t += smem[i];
    __syncthreads();
    return t;
}

// ------------------------------------------------------------ prep kernels ---
__global__ void gat_feat2h(const float* __restrict__ feat,
                           _Float16* __restrict__ featH, int total) {
    int i = blockIdx.x * blockDim.x + threadIdx.x;
    if (i < total) featH[i] = (_Float16)feat[i];
}

__global__ void gat_pack(const float* __restrict__ Wq, const float* __restrict__ Wk,
                         const float* __restrict__ Wv, const float* __restrict__ Ws,
                         const float* __restrict__ bq, const float* __restrict__ bk,
                         const float* __restrict__ bv, const float* __restrict__ bs,
                         _Float16* __restrict__ WallH, float* __restrict__ ball) {
    int i = blockIdx.x * blockDim.x + threadIdx.x;
    if (i < 4 * HD * F_IN) {
        int mat = i >> 16;              // HD*F_IN == 65536
        int rem = i & 65535;
        const float* W = (mat == 0) ? Wq : (mat == 1) ? Wk : (mat == 2) ? Wv : Ws;
        WallH[i] = (_Float16)W[rem];
    }
    if (i < 4 * HD) {
        int mat = i >> 8;
        int rem = i & 255;
        const float* b = (mat == 0) ? bq : (mat == 1) ? bk : (mat == 2) ? bv : bs;
        ball[i] = b[rem];
    }
}

__global__ void gat_init(float* __restrict__ mBuf, float* __restrict__ denom,
                         float* __restrict__ agg, int N) {
    int i = blockIdx.x * blockDim.x + threadIdx.x;
    if (i < N * HD) agg[i] = 0.0f;
    if (i < N * NH) { mBuf[i] = -1e30f; denom[i] = 0.0f; }
}

// ------------------------------------------------- fused projection (WMMA) ---
// qkvs[n, 0:256]=q, [256:512]=k, [512:768]=v, [768:1024]=skip
// grid = ((N+15)/16, 2), block = 256 (8 waves); wave -> 16 rows x 64 cols
// Software-pipelined: fragments for K-step k+1 are loaded (double-buffered in
// VGPRs) before the four WMMAs of step k issue, so the wave never waits
// loadcnt down to zero between consecutive v_wmma ops.
__global__ __launch_bounds__(256) void gat_qkvs_gemm(
        const _Float16* __restrict__ featH, const _Float16* __restrict__ WallH,
        const float* __restrict__ ball, float* __restrict__ qkvs, int N) {
    const int lane    = threadIdx.x & 31;
    const int wave    = threadIdx.x >> 5;
    const int rowBase = blockIdx.x * 16;
    const int colBase = blockIdx.y * 512 + wave * 64;
    const int halfSel = lane >> 4;   // 0 for lanes 0-15, 1 for lanes 16-31
    const int l15     = lane & 15;

    // A fragment addressing (contiguous K order used here, not the ISA table order):
    // lanes 0-15: K = 0..15 ; lanes 16-31: K = 16..31 (contiguous, matches B)
    const _Float16* arow = featH + (size_t)(rowBase + l15) * F_IN + halfSel * 16;
    // B fragment (same contiguous K order as A, not the ISA table order): lane holds one output column
    // (N = l15) and 16 contiguous K values (K base += 16 for lanes 16-31)
    const _Float16* bp[4];
    #pragma unroll
    for (int t = 0; t < 4; t++)
        bp[t] = WallH + (size_t)(colBase + t * 16 + l15) * F_IN + halfSel * 16;

    v16h aBuf[2];
    v16h bBuf[2][4];

    auto loadK = [&](int k0, int buf) {
        v8h alo = *(const v8h*)(arow + k0);
        v8h ahi = *(const v8h*)(arow + k0 + 8);
        #pragma unroll
        for (int i = 0; i < 8; i++) { aBuf[buf][i] = alo[i]; aBuf[buf][8 + i] = ahi[i]; }
        #pragma unroll
        for (int t = 0; t < 4; t++) bBuf[buf][t] = *(const v16h*)(bp[t] + k0);
    };

    const v8f zero = {0.f, 0.f, 0.f, 0.f, 0.f, 0.f, 0.f, 0.f};
    v8f acc[4];
    #pragma unroll
    for (int t = 0; t < 4; t++) acc[t] = zero;

    loadK(0, 0);
    #pragma unroll
    for (int k0 = 0; k0 < F_IN; k0 += 32) {
        const int cur = (k0 >> 5) & 1;
        if (k0 + 32 < F_IN) loadK(k0 + 32, cur ^ 1);   // prefetch next step
        #pragma unroll
        for (int t = 0; t < 4; t++) {
            acc[t] = __builtin_amdgcn_wmma_f32_16x16x32_f16(
                false, aBuf[cur], false, bBuf[cur][t], (short)0, acc[t],
                false, false);
        }
    }

    // D layout: lanes 0-15 hold N=lane, M=0..7 in regs 0..7; lanes 16-31 M=8..15
    const int mBase = halfSel * 8;
    #pragma unroll
    for (int t = 0; t < 4; t++) {
        const int o = colBase + t * 16 + l15;
        const float bias = ball[o];
        #pragma unroll
        for (int r = 0; r < 8; r++) {
            const int row = rowBase + mBase + r;
            if (row < N) qkvs[(size_t)row * KOUT + o] = acc[t][r] + bias;
        }
    }
}

// ------------------------------------------------------------ edge kernels ---
// one wave per edge: 32 lanes x 8 floats cover HD=256; head = lane>>3
__global__ __launch_bounds__(256) void gat_edge_scores(
        const float* __restrict__ qkvs, const int* __restrict__ src,
        const int* __restrict__ dst, float* __restrict__ eBuf,
        float* __restrict__ mBuf, int E) {
    const int lane = threadIdx.x & 31;
    const int edge = blockIdx.x * 8 + (threadIdx.x >> 5);
    if (edge >= E) return;
    const int s = src[edge], d = dst[edge];
    const float4* qp = (const float4*)(qkvs + (size_t)s * KOUT) + lane * 2;
    const float4* kp = (const float4*)(qkvs + (size_t)d * KOUT + HD) + lane * 2;
    float4 q0 = qp[0], q1 = qp[1];
    float4 k0 = kp[0], k1 = kp[1];
    float p = q0.x * k0.x + q0.y * k0.y + q0.z * k0.z + q0.w * k0.w
            + q1.x * k1.x + q1.y * k1.y + q1.z * k1.z + q1.w * k1.w;
    p += __shfl_xor(p, 1, 32);
    p += __shfl_xor(p, 2, 32);
    p += __shfl_xor(p, 4, 32);
    if ((lane & 7) == 0) {
        const int h = lane >> 3;
        const float ev = p * 0.125f;               // 1/sqrt(D=64)
        eBuf[(size_t)edge * NH + h] = ev;
        atomicMaxFloat(mBuf + (size_t)d * NH + h, ev);
    }
}

__global__ void gat_edge_exp(const float* __restrict__ mBuf,
                             const int* __restrict__ dst,
                             float* __restrict__ eBuf,
                             float* __restrict__ denom, int EH) {
    int i = blockIdx.x * blockDim.x + threadIdx.x;
    if (i >= EH) return;
    const int d = dst[i >> 2];
    const int h = i & 3;
    const float ex = __expf(eBuf[i] - mBuf[(size_t)d * NH + h]);
    eBuf[i] = ex;
    atomicAdd(denom + (size_t)d * NH + h, ex);
}

__global__ __launch_bounds__(256) void gat_edge_agg(
        const float* __restrict__ qkvs, const float* __restrict__ eBuf,
        const float* __restrict__ denom, const int* __restrict__ src,
        const int* __restrict__ dst, float* __restrict__ agg, int E) {
    const int lane = threadIdx.x & 31;
    const int edge = blockIdx.x * 8 + (threadIdx.x >> 5);
    if (edge >= E) return;
    const int s = src[edge], d = dst[edge];
    const int h = lane >> 3;
    const float sa = eBuf[(size_t)edge * NH + h] / denom[(size_t)d * NH + h];
    const float* vp = qkvs + (size_t)s * KOUT + 2 * HD + lane * 8;
    float* ap = agg + (size_t)d * HD + lane * 8;
    float4 v0 = *(const float4*)vp;
    float4 v1 = *(const float4*)(vp + 4);
    atomicAdd(ap + 0, sa * v0.x);
    atomicAdd(ap + 1, sa * v0.y);
    atomicAdd(ap + 2, sa * v0.z);
    atomicAdd(ap + 3, sa * v0.w);
    atomicAdd(ap + 4, sa * v1.x);
    atomicAdd(ap + 5, sa * v1.y);
    atomicAdd(ap + 6, sa * v1.z);
    atomicAdd(ap + 7, sa * v1.w);
}

// --------------------------------------------- gate + layernorm + prelu ---
__global__ __launch_bounds__(256) void gat_finalize(
        const float* __restrict__ qkvs, const float* __restrict__ agg,
        const float* __restrict__ Wg, const float* __restrict__ bg,
        const float* __restrict__ lnw, const float* __restrict__ lnb,
        const float* __restrict__ pa, float* __restrict__ out, int N) {
    __shared__ float sred[8];
    const int n = blockIdx.x;
    const int c = threadIdx.x;
    float rstv  = agg[(size_t)n * HD + c];
    float skipv = qkvs[(size_t)n * KOUT + 3 * HD + c];
    // gate: z = Wg[0:256].skip + Wg[256:512].rst + Wg[512:768].(skip-rst)
    const float w1 = Wg[c], w2 = Wg[HD + c], w3 = Wg[2 * HD + c];
    float z = (w1 + w3) * skipv + (w2 - w3) * rstv;
    z = blockReduceSum256(z, sred);
    const float g = 1.0f / (1.0f + __expf(-(z + bg[0])));
    rstv = g * skipv + (1.0f - g) * rstv;
    // layer norm over 256
    const float s1 = blockReduceSum256(rstv, sred);
    const float s2 = blockReduceSum256(rstv * rstv, sred);
    const float mu  = s1 * (1.0f / HD);
    const float var = s2 * (1.0f / HD) - mu * mu;
    float y = (rstv - mu) * rsqrtf(var + 1e-5f) * lnw[c] + lnb[c];
    const float a = pa[0];
    out[(size_t)n * HD + c] = (y >= 0.0f) ? y : a * y;
}

// ------------------------------------------------------------------ launch ---
extern "C" void kernel_launch(void* const* d_in, const int* in_sizes, int n_in,
                              void* d_out, int out_size, void* d_ws, size_t ws_size,
                              hipStream_t stream) {
    const float* feat = (const float*)d_in[0];
    const int*   src  = (const int*)d_in[1];
    const int*   dst  = (const int*)d_in[2];
    const float* Wq = (const float*)d_in[3];
    const float* bq = (const float*)d_in[4];
    const float* Wk = (const float*)d_in[5];
    const float* bk = (const float*)d_in[6];
    const float* Wv = (const float*)d_in[7];
    const float* bv = (const float*)d_in[8];
    const float* Ws = (const float*)d_in[9];
    const float* bs = (const float*)d_in[10];
    const float* Wg = (const float*)d_in[11];
    const float* bg = (const float*)d_in[12];
    const float* lnw = (const float*)d_in[13];
    const float* lnb = (const float*)d_in[14];
    const float* pa  = (const float*)d_in[15];

    const int N = in_sizes[0] / F_IN;
    const int E = in_sizes[1];
    (void)n_in; (void)ws_size; (void)out_size;

    // workspace carve-up (256B aligned slices)
    char* ws = (char*)d_ws;
    size_t off = 0;
    auto carve = [&](size_t bytes) -> char* {
        char* p = ws + off;
        off = (off + bytes + 255) & ~(size_t)255;
        return p;
    };
    _Float16* featH = (_Float16*)carve((size_t)(N + 16) * F_IN * sizeof(_Float16));
    _Float16* WallH = (_Float16*)carve((size_t)4 * HD * F_IN * sizeof(_Float16));
    float*    ball  = (float*)carve((size_t)4 * HD * sizeof(float));
    float*    qkvs  = (float*)carve((size_t)N * KOUT * sizeof(float));
    float*    eBuf  = (float*)carve((size_t)E * NH * sizeof(float));
    float*    mBuf  = (float*)carve((size_t)N * NH * sizeof(float));
    float*    denom = (float*)carve((size_t)N * NH * sizeof(float));
    float*    agg   = (float*)carve((size_t)N * HD * sizeof(float));

    // prep: fp32 -> fp16 conversions, bias pack, accumulator init
    const int totF = N * F_IN;
    gat_feat2h<<<(totF + 255) / 256, 256, 0, stream>>>(feat, featH, totF);
    gat_pack<<<(4 * HD * F_IN + 255) / 256, 256, 0, stream>>>(
        Wq, Wk, Wv, Ws, bq, bk, bv, bs, WallH, ball);
    gat_init<<<(N * HD + 255) / 256, 256, 0, stream>>>(mBuf, denom, agg, N);

    // fused q|k|v|skip projection via WMMA
    dim3 ggrid((N + 15) / 16, KOUT / 512);
    gat_qkvs_gemm<<<ggrid, 256, 0, stream>>>(featH, WallH, ball, qkvs, N);

    // edge phases
    gat_edge_scores<<<(E + 7) / 8, 256, 0, stream>>>(qkvs, src, dst, eBuf, mBuf, E);
    gat_edge_exp<<<(E * NH + 255) / 256, 256, 0, stream>>>(mBuf, dst, eBuf, denom, E * NH);
    gat_edge_agg<<<(E + 7) / 8, 256, 0, stream>>>(qkvs, eBuf, denom, src, dst, agg, E);

    // per-node gate + LN + PReLU
    gat_finalize<<<N, 256, 0, stream>>>(qkvs, agg, Wg, bg, lnw, lnb, pa,
                                        (float*)d_out, N);
}
